// MergedQKVParallelLinearWithDelta_82867099009596
// MI455X (gfx1250) — matrix operand order repaired and checked by reading, not compiled, not run on hardware
//
#include <hip/hip_runtime.h>

#define TOKENS 4096
#define HIDDEN 4096
#define NOUT   6144
#define QSZ    4096
#define KVSZ   1024
#define NADPT  4

typedef __attribute__((ext_vector_type(16))) __bf16 v16bf;
typedef __attribute__((ext_vector_type(8)))  __bf16 v8bf;
typedef __attribute__((ext_vector_type(8)))  float  v8f;

__device__ __forceinline__ unsigned short f2bf(float f) {
  union { float f; unsigned int u; } v; v.f = f;
  unsigned int r = v.u + 0x7FFFu + ((v.u >> 16) & 1u);   // round-to-nearest-even
  return (unsigned short)(r >> 16);
}

__device__ __forceinline__ uint4 pack8(float4 a, float4 b) {
  union { unsigned short h[8]; uint4 v; } u;
  u.h[0] = f2bf(a.x); u.h[1] = f2bf(a.y); u.h[2] = f2bf(a.z); u.h[3] = f2bf(a.w);
  u.h[4] = f2bf(b.x); u.h[5] = f2bf(b.y); u.h[6] = f2bf(b.z); u.h[7] = f2bf(b.w);
  return u.v;
}

__device__ __forceinline__ v16bf cat8(v8bf lo, v8bf hi) {
  return __builtin_shufflevector(lo, hi, 0,1,2,3,4,5,6,7,8,9,10,11,12,13,14,15);
}

// async memory->LDS copy of 16 bytes (gfx1250 GLOBAL_LOAD_ASYNC_TO_LDS_B128, ASYNCcnt)
__device__ __forceinline__ void async_copy_b128(unsigned lds_byte_addr, const void* gptr) {
  asm volatile("global_load_async_to_lds_b128 %0, %1, off"
               :: "v"(lds_byte_addr), "v"((unsigned long long)(uintptr_t)gptr)
               : "memory");
}

// ---------------- setup kernels: route tokens by adapter ----------------

__global__ void k_init_meta(int* meta) {
  if (threadIdx.x < 16) meta[threadIdx.x] = 0;
}

__global__ void k_count(const int* __restrict__ idx, int* __restrict__ meta) {
  int t = blockIdx.x * blockDim.x + threadIdx.x;
  if (t < TOKENS) atomicAdd(&meta[idx[t]], 1);
}

__global__ void k_offsets(int* meta) {
  if (threadIdx.x == 0 && blockIdx.x == 0) {
    int off = 0;
    for (int d = 0; d < NADPT; ++d) {
      meta[4 + d] = off;        // segment start
      meta[8 + d] = off;        // scatter cursor
      off += meta[d];
    }
  }
}

__global__ void k_scatter(const int* __restrict__ idx, int* __restrict__ meta,
                          int* __restrict__ perm) {
  int t = blockIdx.x * blockDim.x + threadIdx.x;
  if (t < TOKENS) {
    int d = idx[t];
    int pos = atomicAdd(&meta[8 + d], 1);
    perm[pos] = t;
  }
}

// ---------------- fp32 -> bf16 conversion ----------------

__global__ void __launch_bounds__(256)
k_convert(const float* __restrict__ src, unsigned short* __restrict__ dst, long n) {
  long i = ((long)blockIdx.x * blockDim.x + threadIdx.x) * 8;
  if (i >= n) return;
  float4 a = *(const float4*)(src + i);
  float4 b = *(const float4*)(src + i + 4);
  *(uint4*)(dst + i) = pack8(a, b);
}

__global__ void __launch_bounds__(256)
k_gather_convert(const float* __restrict__ x, const int* __restrict__ perm,
                 unsigned short* __restrict__ xpb) {
  int g = blockIdx.x;               // permuted row
  int t = perm[g];                  // source token
  const float* src = x + (size_t)t * HIDDEN;
  unsigned short* dst = xpb + (size_t)g * HIDDEN;
  int c0 = threadIdx.x * 16;        // 256 threads * 16 = 4096 cols
  #pragma unroll
  for (int q = 0; q < 2; ++q) {
    float4 a = *(const float4*)(src + c0 + q * 8);
    float4 b = *(const float4*)(src + c0 + q * 8 + 4);
    *(uint4*)(dst + c0 + q * 8) = pack8(a, b);
  }
}

// ---------------- base GEMM: out = x @ W^T  (bf16 WMMA, fp32 acc) ----------------
// block tile 128(M) x 256(N) x 32(K); 8 waves as 2(M) x 4(N); 4x4 WMMA tiles/wave.
// Double-buffered LDS filled by GLOBAL_LOAD_ASYNC_TO_LDS_B128 (ASYNCcnt), so the
// async fill of tile i+1 overlaps the WMMA chain on tile i.

#define LDA 48   // padded LDS row stride (elements); 96B rows keep 16B alignment

__global__ void __launch_bounds__(256)
k_base_gemm(const unsigned short* __restrict__ xb, const unsigned short* __restrict__ wb,
            float* __restrict__ out) {
  __shared__ unsigned short As[2][128 * LDA];
  __shared__ unsigned short Bs[2][256 * LDA];

  const int n0 = blockIdx.x * 256;
  const int m0 = blockIdx.y * 128;
  const int tid   = threadIdx.x;
  const int wid   = tid >> 5;
  const int lane  = tid & 31;
  const int waveM = wid >> 2;      // 0..1  -> 64 rows
  const int waveN = wid & 3;       // 0..3  -> 64 cols
  const int lhalf = lane >> 4;     // 0/1
  const int lrow  = lane & 15;

  v8f acc[4][4];
  #pragma unroll
  for (int mt = 0; mt < 4; ++mt)
    #pragma unroll
    for (int nt = 0; nt < 4; ++nt) acc[mt][nt] = (v8f)(0.f);

  // issue one K-tile's async fill: 6 x b128 per thread (2 for A, 4 for B)
  auto issue_tile = [&](int kk, int p) {
    unsigned ldsA = (unsigned)(uintptr_t)&As[p][0];
    unsigned ldsB = (unsigned)(uintptr_t)&Bs[p][0];
    #pragma unroll
    for (int c = 0; c < 2; ++c) {                 // A: 128 rows x 4 chunks = 512
      int chunk = tid + c * 256;
      int row = chunk >> 2;
      int col = (chunk & 3) * 8;
      async_copy_b128(ldsA + (unsigned)(row * LDA + col) * 2,
                      xb + (size_t)(m0 + row) * HIDDEN + kk + col);
    }
    #pragma unroll
    for (int c = 0; c < 4; ++c) {                 // B: 256 rows x 4 chunks = 1024
      int chunk = tid + c * 256;
      int row = chunk >> 2;
      int col = (chunk & 3) * 8;
      async_copy_b128(ldsB + (unsigned)(row * LDA + col) * 2,
                      wb + (size_t)(n0 + row) * HIDDEN + kk + col);
    }
  };

  const int NK = HIDDEN / 32;      // 128
  issue_tile(0, 0);

  #pragma unroll 1
  for (int ki = 0; ki < NK; ++ki) {
    const int p = ki & 1;
    if (ki + 1 < NK) {
      issue_tile((ki + 1) * 32, p ^ 1);           // fill other buffer (safe: its readers
                                                  // finished before last end-of-iter barrier)
      asm volatile("s_wait_asynccnt 0x6" ::: "memory");  // this wave's tile-ki chunks done
    } else {
      asm volatile("s_wait_asynccnt 0x0" ::: "memory");
    }
    __syncthreads();                              // all waves' fills for buffer p visible

    // fragment loads per ISA 16-bit layouts
    v16bf a[4], b[4];
    #pragma unroll
    for (int mt = 0; mt < 4; ++mt) {
      const unsigned short* fp = &As[p][(waveM * 64 + mt * 16 + lrow) * LDA + lhalf * 8];
      a[mt] = cat8(*(const v8bf*)fp, *(const v8bf*)(fp + 16)); // K {0..7,16..23}/{8..15,24..31}
    }
    #pragma unroll
    for (int nt = 0; nt < 4; ++nt) {
      const unsigned short* fp = &Bs[p][(waveN * 64 + nt * 16 + lrow) * LDA + lhalf * 8];
      b[nt] = cat8(*(const v8bf*)fp, *(const v8bf*)(fp + 16)); // K {0..7,16..23}/{8..15,24..31}
    }
    #pragma unroll
    for (int mt = 0; mt < 4; ++mt)
      #pragma unroll
      for (int nt = 0; nt < 4; ++nt)
        acc[mt][nt] = __builtin_amdgcn_wmma_f32_16x16x32_bf16(
            false, a[mt], false, b[nt], (short)0, acc[mt][nt], false, false);

    __syncthreads();                              // buffer p free for refill next iter
  }

  // C/D layout: lane l, vgpr j -> M = j + 8*(l>=16), N = l&15
  #pragma unroll
  for (int mt = 0; mt < 4; ++mt)
    #pragma unroll
    for (int nt = 0; nt < 4; ++nt) {
      int gn = n0 + waveN * 64 + nt * 16 + lrow;
      #pragma unroll
      for (int j = 0; j < 8; ++j) {
        int gm = m0 + waveM * 64 + mt * 16 + 8 * lhalf + j;
        out[(size_t)gm * NOUT + gn] = acc[mt][nt][j];
      }
    }
}

// ---------------- delta GEMM per adapter segment, int4 dequant on the fly ----
// block tile 128(M) x 128(N) x 32(K); 8 waves as 2(M) x 4(N); 4x2 WMMA tiles/wave.
// A (gathered activations) staged via async-to-LDS with row clamping: rows past
// the segment end load valid-but-unused data; their results are dropped at the
// bounds-checked scatter-add. B is dequantized int4 -> bf16 into LDS.

__global__ void __launch_bounds__(256)
k_delta_gemm(const unsigned short* __restrict__ xpb, const int* __restrict__ perm,
             const int* __restrict__ meta,
             const int* __restrict__ qw_q, const int* __restrict__ qw_k, const int* __restrict__ qw_v,
             const int* __restrict__ qz_q, const int* __restrict__ qz_k, const int* __restrict__ qz_v,
             const float* __restrict__ sc_q, const float* __restrict__ sc_k, const float* __restrict__ sc_v,
             float* __restrict__ out) {
  const int d   = blockIdx.z;
  const int cnt = meta[d];
  const int mb  = blockIdx.y;
  if (mb * 128 >= cnt) return;                // uniform across block
  const int segStart = meta[4 + d];
  const int segEnd   = segStart + cnt;
  const int n0 = blockIdx.x * 128;

  const int* qw; const int* qz; const float* sc; int o0, secRows;
  if (n0 < QSZ)             { qw = qw_q; qz = qz_q; sc = sc_q; o0 = n0;              secRows = QSZ / 8;  }
  else if (n0 < QSZ + KVSZ) { qw = qw_k; qz = qz_k; sc = sc_k; o0 = n0 - QSZ;        secRows = KVSZ / 8; }
  else                      { qw = qw_v; qz = qz_v; sc = sc_v; o0 = n0 - QSZ - KVSZ; secRows = KVSZ / 8; }
  const int p0 = o0 >> 3;

  __shared__ unsigned short As[128 * LDA];
  __shared__ unsigned short Bs[128 * LDA];

  const int tid   = threadIdx.x;
  const int wid   = tid >> 5;
  const int lane  = tid & 31;
  const int waveM = wid >> 2;      // 0..1 -> 64 rows
  const int waveN = wid & 3;       // 0..3 -> 32 cols
  const int lhalf = lane >> 4;
  const int lrow  = lane & 15;

  const int pq = tid >> 5;         // packed-row pair base: handles rows pq and pq+8
  const int hq = tid & 31;         // k-column within tile: 0..31
  const size_t qzBase = (size_t)d * (HIDDEN / 8);
  const size_t scBase = (size_t)d * HIDDEN;

  v8f acc[4][2];
  #pragma unroll
  for (int mt = 0; mt < 4; ++mt)
    #pragma unroll
    for (int nt = 0; nt < 2; ++nt) acc[mt][nt] = (v8f)(0.f);

  const unsigned ldsA = (unsigned)(uintptr_t)&As[0];

  #pragma unroll 1
  for (int kk = 0; kk < HIDDEN; kk += 32) {
    // A: 128 rows x 32 cols bf16, async fill, clamp rows past segment end
    // (safe: previous end-of-iter barrier means no wave still reads As)
    #pragma unroll
    for (int c = 0; c < 2; ++c) {
      int chunk = tid + c * 256;                  // 128 rows x 4 chunks = 512
      int row = chunk >> 2;
      int col = (chunk & 3) * 8;
      int g = segStart + mb * 128 + row;
      if (g >= segEnd) g = segEnd - 1;
      async_copy_b128(ldsA + (unsigned)(row * LDA + col) * 2,
                      xpb + (size_t)g * HIDDEN + kk + col);
    }
    // B: dequantize int4 -> bf16 into LDS; each thread expands 2 packed int32
    // (16 output rows) at its k-column
    {
      int h = kk + hq;
      unsigned int z8 = (unsigned int)qz[qzBase + (h >> 3)];
      float z = (float)((z8 >> ((h & 7) * 4)) & 0xFu);
      float s = sc[scBase + h];
      #pragma unroll
      for (int c = 0; c < 2; ++c) {
        int prow = pq + c * 8;                    // packed row 0..15 within tile
        unsigned int w8 = (unsigned int)qw[((size_t)d * secRows + p0 + prow) * HIDDEN + h];
        #pragma unroll
        for (int j = 0; j < 8; ++j) {
          float val = ((float)((w8 >> (4 * j)) & 0xFu) - z) * s;
          Bs[(prow * 8 + j) * LDA + hq] = f2bf(val);
        }
      }
    }
    asm volatile("s_wait_asynccnt 0x0" ::: "memory");  // own A chunks landed
    __syncthreads();                                   // + all waves' A/B visible

    v16bf a[4], b[2];
    #pragma unroll
    for (int mt = 0; mt < 4; ++mt) {
      const unsigned short* fp = &As[(waveM * 64 + mt * 16 + lrow) * LDA + lhalf * 8];
      a[mt] = cat8(*(const v8bf*)fp, *(const v8bf*)(fp + 16));
    }
    #pragma unroll
    for (int nt = 0; nt < 2; ++nt) {
      const unsigned short* fp = &Bs[(waveN * 32 + nt * 16 + lrow) * LDA + lhalf * 8];
      b[nt] = cat8(*(const v8bf*)fp, *(const v8bf*)(fp + 16));
    }
    #pragma unroll
    for (int mt = 0; mt < 4; ++mt)
      #pragma unroll
      for (int nt = 0; nt < 2; ++nt)
        acc[mt][nt] = __builtin_amdgcn_wmma_f32_16x16x32_bf16(
            false, a[mt], false, b[nt], (short)0, acc[mt][nt], false, false);
    __syncthreads();                                   // LDS free for next iter
  }

  // scatter-add into out rows via perm (rows unique per token -> plain RMW)
  #pragma unroll
  for (int mt = 0; mt < 4; ++mt)
    #pragma unroll
    for (int nt = 0; nt < 2; ++nt) {
      int gn = n0 + waveN * 32 + nt * 16 + lrow;
      int rbase = waveM * 64 + mt * 16 + 8 * lhalf;
      #pragma unroll
      for (int j = 0; j < 8; ++j) {
        int g = segStart + mb * 128 + rbase + j;
        if (g < segEnd) {
          int t = perm[g];
          float* p = &out[(size_t)t * NOUT + gn];
          *p = *p + acc[mt][nt][j];
        }
      }
    }
}

// ---------------- launch ----------------

extern "C" void kernel_launch(void* const* d_in, const int* in_sizes, int n_in,
                              void* d_out, int out_size, void* d_ws, size_t ws_size,
                              hipStream_t stream) {
  (void)in_sizes; (void)n_in; (void)out_size; (void)ws_size;
  const float* x       = (const float*)d_in[0];
  const int*   indices = (const int*)d_in[1];
  const float* W       = (const float*)d_in[2];
  const int*   qw_q    = (const int*)d_in[3];
  const int*   qw_k    = (const int*)d_in[4];
  const int*   qw_v    = (const int*)d_in[5];
  const int*   qz_q    = (const int*)d_in[6];
  const int*   qz_k    = (const int*)d_in[7];
  const int*   qz_v    = (const int*)d_in[8];
  const float* sc_q    = (const float*)d_in[9];
  const float* sc_k    = (const float*)d_in[10];
  const float* sc_v    = (const float*)d_in[11];
  float* out = (float*)d_out;

  char* ws = (char*)d_ws;
  size_t off = 0;
  int* meta = (int*)(ws + off);            off += 256;
  int* perm = (int*)(ws + off);            off += (size_t)TOKENS * sizeof(int);
  off = (off + 255) & ~(size_t)255;
  unsigned short* xb  = (unsigned short*)(ws + off); off += (size_t)TOKENS * HIDDEN * 2;
  unsigned short* xpb = (unsigned short*)(ws + off); off += (size_t)TOKENS * HIDDEN * 2;
  unsigned short* wb  = (unsigned short*)(ws + off); off += (size_t)NOUT   * HIDDEN * 2;

  k_init_meta<<<1, 32, 0, stream>>>(meta);
  k_count<<<TOKENS / 256, 256, 0, stream>>>(indices, meta);
  k_offsets<<<1, 1, 0, stream>>>(meta);
  k_scatter<<<TOKENS / 256, 256, 0, stream>>>(indices, meta, perm);

  long nx = (long)TOKENS * HIDDEN;
  long nw = (long)NOUT * HIDDEN;
  k_convert<<<(int)((nx / 8 + 255) / 256), 256, 0, stream>>>(x, xb, nx);
  k_convert<<<(int)((nw / 8 + 255) / 256), 256, 0, stream>>>(W, wb, nw);
  k_gather_convert<<<TOKENS, 256, 0, stream>>>(x, perm, xpb);

  dim3 gb(NOUT / 256, TOKENS / 128);
  k_base_gemm<<<gb, 256, 0, stream>>>(xb, wb, out);

  dim3 gd(NOUT / 128, TOKENS / 128, NADPT);
  k_delta_gemm<<<gd, 256, 0, stream>>>(xpb, perm, meta,
                                       qw_q, qw_k, qw_v,
                                       qz_q, qz_k, qz_v,
                                       sc_q, sc_k, sc_v, out);
}
